// MultiHeadSelfAttention_78623671320875
// MI455X (gfx1250) — matrix operand order repaired and checked by reading, not compiled, not run on hardware
//
#include <hip/hip_runtime.h>

typedef __bf16 bf16_t;
typedef __attribute__((ext_vector_type(16))) __bf16 v16bf;
typedef __attribute__((ext_vector_type(8)))  __bf16 v8bf;
typedef __attribute__((ext_vector_type(8)))  float  v8f;

#define WMMA_BF16(A, B, C) \
  __builtin_amdgcn_wmma_f32_16x16x32_bf16(false, (A), false, (B), (short)0, (C), false, false)

// ---------------------------------------------------------------------------
// Fragment helpers (wave32; A/B use this code's contiguous k order, not the ISA table order):
//  A 16x32 bf16: lane L holds row r=L&15; elems i  = K=16h+i (h=L>>4)
//  B 32x16 bf16: lane L holds col c=L&15; elems i  = K=16h+i
//  C 16x16 f32 : lane L, vgpr g -> (M = g+8h, N = L&15)
// ---------------------------------------------------------------------------

__device__ __forceinline__ v16bf frag_a_f32g(const float* __restrict__ row, int h) {
  v16bf f;
  const float* lo = row + 16 * h;
  const float* hi = row + 16 * h + 8;
#pragma unroll
  for (int i = 0; i < 8; ++i) f[i] = (__bf16)lo[i];
#pragma unroll
  for (int i = 0; i < 8; ++i) f[8 + i] = (__bf16)hi[i];
  return f;
}

__device__ __forceinline__ v16bf frag_a_bf16g(const bf16_t* __restrict__ row, int h) {
  v8bf lo = *(const v8bf*)(row + 16 * h);
  v8bf hi = *(const v8bf*)(row + 16 * h + 8);
  v16bf f;
#pragma unroll
  for (int i = 0; i < 8; ++i) { f[i] = lo[i]; f[8 + i] = hi[i]; }
  return f;
}

// stage one 16-float chunk -> 16 bf16 in LDS (32B store)
__device__ __forceinline__ void stage16_f32(const float* __restrict__ src, bf16_t* __restrict__ dst) {
  v16bf t;
#pragma unroll
  for (int i = 0; i < 16; ++i) t[i] = (__bf16)src[i];
  *(v16bf*)dst = t;
}

// ---------------------------------------------------------------------------
// Projection producing transposed output tiles (used for Q and K):
//   C^T[n][m] = sum_k W[n][k] * X[m][k]   (A' = W rows, B' = X rows via LDS)
// Output: head-split [B=4][H=16][S=2048][DK=64] bf16, scaled by alpha.
// Double-buffered LDS staging: one barrier per K-step, next tile's global
// loads overlap current tile's WMMAs.
// ---------------------------------------------------------------------------
__global__ __launch_bounds__(128) void proj_qk_kernel(
    const float* __restrict__ X, const float* __restrict__ W,
    const float* __restrict__ bias, bf16_t* __restrict__ out,
    float alpha, int Mtiles) {
  __shared__ bf16_t lX[2 * 64 * 32];
  const int tile = blockIdx.x;
  const int tM = tile % Mtiles, tN = tile / Mtiles;
  const int m0 = tM * 64, n0 = tN * 64;
  const int tid = threadIdx.x, w = tid >> 5, lane = tid & 31;
  const int c = lane & 15, h = lane >> 4;
  const int sr = tid >> 1, sp = (tid & 1) * 16;

  v8f acc[4] = {};
  const float* wrow = W + (size_t)(n0 + w * 16 + c) * 1024;
  const float* xsrc = X + (size_t)(m0 + sr) * 1024 + sp;

  stage16_f32(xsrc, lX + sr * 32 + sp);
  __syncthreads();

  int p = 0;
  for (int k0 = 0; k0 < 1024; k0 += 32) {
    const bf16_t* cur = lX + p * 2048;
    if (k0 + 32 < 1024)
      stage16_f32(xsrc + k0 + 32, lX + (p ^ 1) * 2048 + sr * 32 + sp);
    const v16bf fa  = frag_a_f32g(wrow + k0, h);
    const v16bf fb0 = *(const v16bf*)(cur + (0 * 16 + c) * 32 + 16 * h);
    const v16bf fb1 = *(const v16bf*)(cur + (1 * 16 + c) * 32 + 16 * h);
    const v16bf fb2 = *(const v16bf*)(cur + (2 * 16 + c) * 32 + 16 * h);
    const v16bf fb3 = *(const v16bf*)(cur + (3 * 16 + c) * 32 + 16 * h);
    acc[0] = WMMA_BF16(fa, fb0, acc[0]);
    acc[1] = WMMA_BF16(fa, fb1, acc[1]);
    acc[2] = WMMA_BF16(fa, fb2, acc[2]);
    acc[3] = WMMA_BF16(fa, fb3, acc[3]);
    __syncthreads();
    p ^= 1;
  }

  float bvv[8];
  const float* bp = bias + n0 + w * 16 + 8 * h;
#pragma unroll
  for (int g = 0; g < 8; ++g) bvv[g] = bp[g];

  const int hd = tN;
#pragma unroll
  for (int t = 0; t < 4; ++t) {
    const int m = m0 + t * 16 + c;
    const int b = m >> 11, s = m & 2047;
    bf16_t* dst = out + (((size_t)b * 16 + hd) * 2048 + s) * 64 + w * 16 + 8 * h;
    v8bf v;
#pragma unroll
    for (int g = 0; g < 8; ++g) v[g] = (__bf16)((acc[t][g] + bvv[g]) * alpha);
    *(v8bf*)dst = v;
  }
}

// ---------------------------------------------------------------------------
// V projection, normal orientation: C[m][n] = sum_k X[m][k] W[n][k]
//   (A = X rows, B = W rows via LDS). Output stored transposed d-major:
//   Vt [B][H][DK=64][S=2048] bf16  (lane owns fixed d, 8 contiguous s).
// ---------------------------------------------------------------------------
__global__ __launch_bounds__(128) void proj_v_kernel(
    const float* __restrict__ X, const float* __restrict__ W,
    const float* __restrict__ bias, bf16_t* __restrict__ Vt, int Mtiles) {
  __shared__ bf16_t lW[2 * 64 * 32];
  const int tile = blockIdx.x;
  const int tM = tile % Mtiles, tN = tile / Mtiles;
  const int m0 = tM * 64, n0 = tN * 64;
  const int tid = threadIdx.x, w = tid >> 5, lane = tid & 31;
  const int c = lane & 15, h = lane >> 4;
  const int sr = tid >> 1, sp = (tid & 1) * 16;

  v8f acc[4] = {};
  const float* xrow = X + (size_t)(m0 + w * 16 + c) * 1024;
  const float* wsrc = W + (size_t)(n0 + sr) * 1024 + sp;

  stage16_f32(wsrc, lW + sr * 32 + sp);
  __syncthreads();

  int p = 0;
  for (int k0 = 0; k0 < 1024; k0 += 32) {
    const bf16_t* cur = lW + p * 2048;
    if (k0 + 32 < 1024)
      stage16_f32(wsrc + k0 + 32, lW + (p ^ 1) * 2048 + sr * 32 + sp);
    const v16bf fa  = frag_a_f32g(xrow + k0, h);
    const v16bf fb0 = *(const v16bf*)(cur + (0 * 16 + c) * 32 + 16 * h);
    const v16bf fb1 = *(const v16bf*)(cur + (1 * 16 + c) * 32 + 16 * h);
    const v16bf fb2 = *(const v16bf*)(cur + (2 * 16 + c) * 32 + 16 * h);
    const v16bf fb3 = *(const v16bf*)(cur + (3 * 16 + c) * 32 + 16 * h);
    acc[0] = WMMA_BF16(fa, fb0, acc[0]);
    acc[1] = WMMA_BF16(fa, fb1, acc[1]);
    acc[2] = WMMA_BF16(fa, fb2, acc[2]);
    acc[3] = WMMA_BF16(fa, fb3, acc[3]);
    __syncthreads();
    p ^= 1;
  }

  const int hd = tN;
  const int mrow = m0 + w * 16 + 8 * h;       // first of 8 consecutive m (same b)
  const int b = mrow >> 11, s = mrow & 2047;
#pragma unroll
  for (int t = 0; t < 4; ++t) {
    const int n = n0 + t * 16 + c;
    const float bvs = bias[n];
    const int d = t * 16 + c;
    bf16_t* dst = Vt + (((size_t)b * 16 + hd) * 64 + d) * 2048 + s;
    v8bf v;
#pragma unroll
    for (int g = 0; g < 8; ++g) v[g] = (__bf16)(acc[t][g] + bvs);
    *(v8bf*)dst = v;
  }
}

// ---------------------------------------------------------------------------
// Flash attention. One block = 4 waves = 64 queries of one (b,h).
// Wave computes T = K_blk * Q^T (lane = query column), streaming softmax,
// then out^T = V^T * P^T accumulated over 2048 keys in 32-key steps.
// Q was pre-scaled by 1/8 in the projection. V fragments are loaded before
// the softmax VALU work so their latency hides under exp/max/shuffles.
// ---------------------------------------------------------------------------
__global__ __launch_bounds__(128) void attn_kernel(
    const bf16_t* __restrict__ Q, const bf16_t* __restrict__ K,
    const bf16_t* __restrict__ Vt, bf16_t* __restrict__ AO) {
  const int blk = blockIdx.x;      // B*H*(S/64) = 2048
  const int sblk = blk & 31;       // S/64 = 32
  const int bh = blk >> 5;         // b*16 + h
  const int tid = threadIdx.x, w = tid >> 5, lane = tid & 31;
  const int c = lane & 15, h = lane >> 4;
  const bool upper = (h != 0);

  const bf16_t* Qb = Q + (size_t)bh * 2048 * 64;
  const bf16_t* Kb = K + (size_t)bh * 2048 * 64;
  const bf16_t* Vb = Vt + (size_t)bh * 64 * 2048;

  const int q = sblk * 64 + w * 16 + c;
  const v16bf qb0 = *(const v16bf*)(Qb + (size_t)q * 64 + 16 * h);
  const v16bf qb1 = *(const v16bf*)(Qb + (size_t)q * 64 + 32 + 16 * h);

  v8f acc0 = {}, acc1 = {}, acc2 = {}, acc3 = {};
  float m_run = -3.0e38f, l_run = 0.0f;

  for (int j = 0; j < 2048; j += 32) {
    // ---- K fragments (all four issued before the score WMMAs)
    const bf16_t* kr  = Kb + (size_t)(j + c) * 64;
    const bf16_t* kr2 = Kb + (size_t)(j + 16 + c) * 64;
    const v16bf ka0 = frag_a_bf16g(kr, h);
    const v16bf ka1 = frag_a_bf16g(kr + 32, h);
    const v16bf ka2 = frag_a_bf16g(kr2, h);
    const v16bf ka3 = frag_a_bf16g(kr2 + 32, h);

    v8f t0 = {}, t1 = {};
    t0 = WMMA_BF16(ka0, qb0, t0);
    t0 = WMMA_BF16(ka1, qb1, t0);
    t1 = WMMA_BF16(ka2, qb0, t1);
    t1 = WMMA_BF16(ka3, qb1, t1);

    // ---- V fragments: independent of softmax; issue loads now
    const bf16_t* vr = Vb + (size_t)c * 2048 + j;
    const v16bf va0 = frag_a_bf16g(vr + (size_t)0 * 16 * 2048, h);
    const v16bf va1 = frag_a_bf16g(vr + (size_t)1 * 16 * 2048, h);
    const v16bf va2 = frag_a_bf16g(vr + (size_t)2 * 16 * 2048, h);
    const v16bf va3 = frag_a_bf16g(vr + (size_t)3 * 16 * 2048, h);

    if (j + 32 < 2048) {
      __builtin_prefetch(Kb + (size_t)(j + 32 + c) * 64, 0, 0);
      __builtin_prefetch(Vb + (size_t)c * 2048 + j + 32, 0, 0);
    }

    // ---- online softmax (lane owns query c; partner lane L^16 holds other 8 keys)
    float mb = t0[0];
#pragma unroll
    for (int g = 1; g < 8; ++g) mb = fmaxf(mb, t0[g]);
#pragma unroll
    for (int g = 0; g < 8; ++g) mb = fmaxf(mb, t1[g]);
    mb = fmaxf(mb, __shfl_xor(mb, 16, 32));
    const float m_new = fmaxf(m_run, mb);
    const float scale = __expf(m_run - m_new);
    float p0[8], p1[8], lb = 0.0f;
#pragma unroll
    for (int g = 0; g < 8; ++g) {
      p0[g] = __expf(t0[g] - m_new);
      p1[g] = __expf(t1[g] - m_new);
      lb += p0[g] + p1[g];
    }
    lb += __shfl_xor(lb, 16, 32);
    l_run = l_run * scale + lb;
    m_run = m_new;
#pragma unroll
    for (int g = 0; g < 8; ++g) {
      acc0[g] *= scale; acc1[g] *= scale; acc2[g] *= scale; acc3[g] *= scale;
    }

    // ---- repack P^T C-fragments -> B-fragment (lane half h covers keys 16h..16h+15)
    v16bf pb;
#pragma unroll
    for (int g = 0; g < 8; ++g) {
      const float send = upper ? p0[g] : p1[g];
      const float recv = __shfl_xor(send, 16, 32);
      const float lo = upper ? recv : p0[g];   // keys 16h + g
      const float hi = upper ? p1[g] : recv;   // keys 16h + 8 + g
      pb[g] = (__bf16)lo;
      pb[8 + g] = (__bf16)hi;
    }

    // ---- out^T += V^T * P^T  (4 d-tiles)
    acc0 = WMMA_BF16(va0, pb, acc0);
    acc1 = WMMA_BF16(va1, pb, acc1);
    acc2 = WMMA_BF16(va2, pb, acc2);
    acc3 = WMMA_BF16(va3, pb, acc3);
  }

  const float inv_l = 1.0f / l_run;
  const int b = bh >> 4, hd = bh & 15;
  const int s = sblk * 64 + w * 16 + c;
  bf16_t* dst = AO + ((size_t)b * 2048 + s) * 1024 + hd * 64 + 8 * h;

  auto store_tile = [&](const v8f& a, int dt) {
    v8bf v;
#pragma unroll
    for (int g = 0; g < 8; ++g) v[g] = (__bf16)(a[g] * inv_l);
    *(v8bf*)(dst + dt * 16) = v;
  };
  store_tile(acc0, 0);
  store_tile(acc1, 1);
  store_tile(acc2, 2);
  store_tile(acc3, 3);
}

// ---------------------------------------------------------------------------
// Output projection: y^T[n][m] = sum_k wo[n][k] * AO[m][k] + bo[n]
// A' = wo rows (fp32 global), B' = AO rows (bf16, LDS-staged, double-buffered).
// ---------------------------------------------------------------------------
__global__ __launch_bounds__(128) void proj_o_kernel(
    const bf16_t* __restrict__ AO, const float* __restrict__ W,
    const float* __restrict__ bias, float* __restrict__ out, int Mtiles) {
  __shared__ bf16_t lA[2 * 64 * 32];
  const int tile = blockIdx.x;
  const int tM = tile % Mtiles, tN = tile / Mtiles;
  const int m0 = tM * 64, n0 = tN * 64;
  const int tid = threadIdx.x, w = tid >> 5, lane = tid & 31;
  const int c = lane & 15, h = lane >> 4;
  const int sr = tid >> 1, sp = (tid & 1) * 16;

  v8f acc[4] = {};
  const float* wrow = W + (size_t)(n0 + w * 16 + c) * 1024;
  const bf16_t* asrc = AO + (size_t)(m0 + sr) * 1024 + sp;

  *(v16bf*)(lA + sr * 32 + sp) = *(const v16bf*)asrc;
  __syncthreads();

  int p = 0;
  for (int k0 = 0; k0 < 1024; k0 += 32) {
    const bf16_t* cur = lA + p * 2048;
    if (k0 + 32 < 1024)
      *(v16bf*)(lA + (p ^ 1) * 2048 + sr * 32 + sp) = *(const v16bf*)(asrc + k0 + 32);
    const v16bf fa  = frag_a_f32g(wrow + k0, h);
    const v16bf fb0 = *(const v16bf*)(cur + (0 * 16 + c) * 32 + 16 * h);
    const v16bf fb1 = *(const v16bf*)(cur + (1 * 16 + c) * 32 + 16 * h);
    const v16bf fb2 = *(const v16bf*)(cur + (2 * 16 + c) * 32 + 16 * h);
    const v16bf fb3 = *(const v16bf*)(cur + (3 * 16 + c) * 32 + 16 * h);
    acc[0] = WMMA_BF16(fa, fb0, acc[0]);
    acc[1] = WMMA_BF16(fa, fb1, acc[1]);
    acc[2] = WMMA_BF16(fa, fb2, acc[2]);
    acc[3] = WMMA_BF16(fa, fb3, acc[3]);
    __syncthreads();
    p ^= 1;
  }

  float bvv[8];
  const float* bp = bias + n0 + w * 16 + 8 * h;
#pragma unroll
  for (int g = 0; g < 8; ++g) bvv[g] = bp[g];

#pragma unroll
  for (int t = 0; t < 4; ++t) {
    const int m = m0 + t * 16 + c;
    float* dst = out + (size_t)m * 1024 + n0 + w * 16 + 8 * h;
    float r[8];
#pragma unroll
    for (int g = 0; g < 8; ++g) r[g] = acc[t][g] + bvv[g];
    ((float4*)dst)[0] = make_float4(r[0], r[1], r[2], r[3]);
    ((float4*)dst)[1] = make_float4(r[4], r[5], r[6], r[7]);
  }
}

// ---------------------------------------------------------------------------
extern "C" void kernel_launch(void* const* d_in, const int* in_sizes, int n_in,
                              void* d_out, int out_size, void* d_ws, size_t ws_size,
                              hipStream_t stream) {
  (void)in_sizes; (void)n_in; (void)out_size; (void)ws_size;
  const float* x  = (const float*)d_in[0];
  const float* wq = (const float*)d_in[1];
  const float* bq = (const float*)d_in[2];
  const float* wk = (const float*)d_in[3];
  const float* bk = (const float*)d_in[4];
  const float* wv = (const float*)d_in[5];
  const float* bv = (const float*)d_in[6];
  const float* wo = (const float*)d_in[7];
  const float* bo = (const float*)d_in[8];
  float* out = (float*)d_out;

  char* ws = (char*)d_ws;
  const size_t SEG = (size_t)4 * 16 * 2048 * 64 * sizeof(bf16_t);  // 16 MB
  bf16_t* Qh = (bf16_t*)(ws);
  bf16_t* Kh = (bf16_t*)(ws + SEG);
  bf16_t* Vt = (bf16_t*)(ws + 2 * SEG);
  bf16_t* AO = (bf16_t*)(ws + 3 * SEG);

  const int Mtiles = 8192 / 64;          // 128
  const dim3 blk(128);
  const dim3 gproj(Mtiles * 16);         // 2048 blocks
  const dim3 gattn(4 * 16 * 32);         // 2048 blocks

  proj_qk_kernel<<<gproj, blk, 0, stream>>>(x, wq, bq, Qh, 0.125f, Mtiles);
  proj_qk_kernel<<<gproj, blk, 0, stream>>>(x, wk, bk, Kh, 1.0f, Mtiles);
  proj_v_kernel<<<gproj, blk, 0, stream>>>(x, wv, bv, Vt, Mtiles);
  attn_kernel<<<gattn, blk, 0, stream>>>(Qh, Kh, Vt, AO);
  proj_o_kernel<<<gproj, blk, 0, stream>>>(AO, wo, bo, out, Mtiles);
}
